// ConvNN_31671088841450
// MI455X (gfx1250) — matrix operand order repaired and checked by reading, not compiled, not run on hardware
//
#include <hip/hip_runtime.h>
#include <stdint.h>

// ---------------------------------------------------------------------------
// ConvNN pipeline for MI455X (gfx1250, wave32, WMMA 16x16x32 bf16, TDM)
//   x(B,C,N) --1x1 convs + fused l2norm--> kn,qn (bf16), v (f32)
//   S = kn^T qn (WMMA, fused running top-9 + softmax, S never materialized)
//   out = grouped conv == GEMM over K-dim 9*256, gather folded into B-frags,
//         conv weights staged to LDS via tensor_load_to_lds (TDM)
// B=16, C=O=256, N=2048, K=9
// ---------------------------------------------------------------------------

typedef __attribute__((ext_vector_type(16))) __bf16 v16bf;
typedef __attribute__((ext_vector_type(8)))  float  v8f;
typedef __attribute__((ext_vector_type(4)))  unsigned int u32x4;
typedef __attribute__((ext_vector_type(8)))  unsigned int u32x8;
typedef __attribute__((ext_vector_type(4)))  float f32x4;

#define B_ 16
#define C_ 256
#define N_ 2048
#define K_ 9
#define KK_ (K_ * C_)   // 2304 flattened (t,c) contraction dim

union BF16Frag { v16bf v; unsigned short h[16]; u32x4 q[2]; };
union F32Acc   { v8f v; float f[8]; f32x4 q[2]; };

// float -> bf16 round-to-nearest-even, raw u16 storage
__device__ __forceinline__ unsigned short f2bf(float f) {
  unsigned int u = __float_as_uint(f);
  u += 0x7fffu + ((u >> 16) & 1u);
  return (unsigned short)(u >> 16);
}

// A-matrix 16x32 bf16 fragment (this code's k order, not the ISA table; B matches): lane m = lane&15;
// lanes 0-15 hold K = 0..15, lanes 16-31 hold K = 16..31 (contiguous).
__device__ __forceinline__ v16bf load_a_frag(const unsigned short* base, int ld,
                                             int m0, int k0, int lane) {
  int m  = m0 + (lane & 15);
  int kh = (lane >> 4) << 4;               // 0 or 16
  const unsigned short* p = base + (size_t)m * ld + k0 + kh;
  BF16Frag f;
  f.q[0] = *(const u32x4*)(p);
  f.q[1] = *(const u32x4*)(p + 8);
  return f.v;
}

// B-matrix 32x16 bf16 fragment: lane n = lane&15;
// lanes 0-15 hold K=0..15, lanes 16-31 hold K=16..31 (contiguous).
__device__ __forceinline__ v16bf load_b_frag(const unsigned short* base, int ld,
                                             int n0, int k0, int lane) {
  int n  = n0 + (lane & 15);
  int kh = (lane >> 4) << 4;               // 0 or 16
  const unsigned short* p = base + (size_t)n * ld + k0 + kh;
  BF16Frag f;
  f.q[0] = *(const u32x4*)(p);
  f.q[1] = *(const u32x4*)(p + 8);
  return f.v;
}

// A-fragment from an LDS-resident 16x256 bf16 tile (row-major) -> ds_load_b128
__device__ __forceinline__ v16bf lds_a_frag(const unsigned short* tile, int k0, int lane) {
  int m  = lane & 15;
  int kh = (lane >> 4) << 4;
  const unsigned short* p = tile + m * C_ + k0 + kh;
  BF16Frag f;
  f.q[0] = *(const u32x4*)(p);
  f.q[1] = *(const u32x4*)(p + 8);
  return f.v;
}

// --------------------------- prep kernels ----------------------------------
__global__ void prep_x(const float* __restrict__ x, unsigned short* __restrict__ x_t) {
  int idx = blockIdx.x * blockDim.x + threadIdx.x;      // B*C*N
  int n = idx % N_;
  int t = idx / N_;
  int c = t % C_;
  int b = t / C_;
  x_t[((size_t)b * N_ + n) * C_ + c] = f2bf(x[idx]);
}

__global__ void prep_w(const float* __restrict__ wk, const float* __restrict__ wq,
                       const float* __restrict__ wv, unsigned short* __restrict__ w_bf) {
  int idx = blockIdx.x * blockDim.x + threadIdx.x;      // 3*C*C
  int e = idx & (C_ * C_ - 1);
  int which = idx / (C_ * C_);
  const float* s = (which == 0) ? wk : (which == 1) ? wq : wv;
  w_bf[idx] = f2bf(s[e]);
}

// conv_w (O,C,K) f32 -> cw2 (O, K*C) bf16 with kk = t*C + c
__global__ void prep_cw(const float* __restrict__ cw, unsigned short* __restrict__ cw2) {
  int idx = blockIdx.x * blockDim.x + threadIdx.x;      // O*C*K
  int t = idx % K_;
  int r = idx / K_;
  int c = r % C_;
  int o = r / C_;
  cw2[(size_t)o * KK_ + t * C_ + c] = f2bf(cw[idx]);
}

// ---------------- fused QKV GEMM + L2-normalize (k,q) ----------------------
// One wave per (b,which,ntile) computes the full 16-position x 256-channel
// block in registers (16 C-fragments), so k/q are normalized in-register and
// only bf16 normalized rows (and f32 v) ever reach HBM. Saves ~128 MB of
// f32 round-trip traffic (~5.5us at 23.3 TB/s).
__global__ void __launch_bounds__(32) qkv_norm(const unsigned short* __restrict__ x_t,
                                               const unsigned short* __restrict__ w_bf,
                                               const float* __restrict__ bk,
                                               const float* __restrict__ bq,
                                               const float* __restrict__ bv,
                                               unsigned short* __restrict__ kn,
                                               unsigned short* __restrict__ qn,
                                               float* __restrict__ vt) {
  int lane = threadIdx.x & 31;
  int wid = blockIdx.x;                     // B*3*(N/16)
  int ntile = wid % (N_ / 16); wid /= (N_ / 16);
  int which = wid % 3;
  int b = wid / 3;
  int n0 = ntile * 16;
  const unsigned short* wmat = w_bf + (size_t)which * C_ * C_;
  const unsigned short* xrow = x_t + (size_t)b * N_ * C_;
  const float* bias = (which == 0) ? bk : (which == 1) ? bq : bv;
  int off = (lane < 16) ? 0 : 8;            // C/D layout: VGPR r holds M = r+off
  int nl  = n0 + (lane & 15);               // this lane's position

  // B-fragments of x are shared by all 16 o-tiles: preload once (64 VGPRs)
  v16bf bfr[8];
  #pragma unroll
  for (int ks = 0; ks < 8; ++ks) bfr[ks] = load_b_frag(xrow, C_, n0, ks * 32, lane);

  F32Acc acc[16];                           // 128 VGPRs: 16 positions x 256 ch
  #pragma unroll
  for (int ot = 0; ot < 16; ++ot) {
    #pragma unroll
    for (int r = 0; r < 8; ++r) acc[ot].f[r] = 0.f;
    #pragma unroll
    for (int ks = 0; ks < 8; ++ks) {
      v16bf a = load_a_frag(wmat, C_, ot * 16, ks * 32, lane);
      acc[ot].v = __builtin_amdgcn_wmma_f32_16x16x32_bf16(false, a, false, bfr[ks],
                                                          (short)0, acc[ot].v, false, false);
    }
    #pragma unroll
    for (int r = 0; r < 8; ++r) acc[ot].f[r] += bias[ot * 16 + off + r];
  }

  if (which == 2) {                         // v: store f32, no normalization
    float* dst = vt + ((size_t)(b * N_) + nl) * C_ + off;
    #pragma unroll
    for (int ot = 0; ot < 16; ++ot) {
      *(f32x4*)(dst + ot * 16)     = acc[ot].q[0];
      *(f32x4*)(dst + ot * 16 + 4) = acc[ot].q[1];
    }
  } else {                                  // k,q: l2norm over channels, bf16
    float s = 0.f;
    #pragma unroll
    for (int ot = 0; ot < 16; ++ot)
      #pragma unroll
      for (int r = 0; r < 8; ++r) s += acc[ot].f[r] * acc[ot].f[r];
    s += __shfl_xor(s, 16, 32);             // partner lane holds other 8 rows
    float sc = 1.0f / fmaxf(sqrtf(s), 1e-12f);
    unsigned short* dst = ((which == 0) ? kn : qn) + ((size_t)(b * N_) + nl) * C_ + off;
    #pragma unroll
    for (int ot = 0; ot < 16; ++ot) {
      union { unsigned short h[8]; u32x4 q; } pk;
      #pragma unroll
      for (int r = 0; r < 8; ++r) pk.h[r] = f2bf(acc[ot].f[r] * sc);
      *(u32x4*)(dst + ot * 16) = pk.q;
    }
  }
}

// ------------------- S = kn^T qn, fused running top-9 + softmax ------------
__global__ void __launch_bounds__(32) s_topk(const unsigned short* __restrict__ kn,
                                             const unsigned short* __restrict__ qn,
                                             float* __restrict__ topw,
                                             int* __restrict__ topi) {
  __shared__ float tile[16][16];
  int lane = threadIdx.x & 31;
  int wid = blockIdx.x;                    // B * N/16
  int ntile = wid % (N_ / 16);
  int b = wid / (N_ / 16);
  int n0 = ntile * 16;
  const unsigned short* krow = kn + (size_t)b * N_ * C_;
  const unsigned short* qrow = qn + (size_t)b * N_ * C_;

  v16bf afr[8];
  #pragma unroll
  for (int ks = 0; ks < 8; ++ks) afr[ks] = load_a_frag(krow, C_, n0, ks * 32, lane);

  float tv[9]; int ti[9];
  #pragma unroll
  for (int t = 0; t < 9; ++t) { tv[t] = -3.0e38f; ti[t] = 0; }

  for (int m0 = 0; m0 < N_; m0 += 16) {
    F32Acc acc;
    #pragma unroll
    for (int r = 0; r < 8; ++r) acc.f[r] = 0.f;
    #pragma unroll
    for (int ks = 0; ks < 8; ++ks) {
      v16bf bfr = load_b_frag(qrow, C_, m0, ks * 32, lane);
      acc.v = __builtin_amdgcn_wmma_f32_16x16x32_bf16(false, afr[ks], false, bfr,
                                                      (short)0, acc.v, false, false);
    }
    int off = (lane < 16) ? 0 : 8;
    int col = lane & 15;
    #pragma unroll
    for (int r = 0; r < 8; ++r) tile[r + off][col] = acc.f[r];
    __syncthreads();
    if (lane < 16) {                        // one row per lane, sorted-9 insert
      #pragma unroll
      for (int j = 0; j < 16; ++j) {
        float s = tile[lane][j];
        if (s > tv[8]) {
          tv[8] = s; ti[8] = m0 + j;
          #pragma unroll
          for (int qq = 8; qq > 0; --qq) {
            if (tv[qq] > tv[qq - 1]) {
              float tf = tv[qq]; tv[qq] = tv[qq - 1]; tv[qq - 1] = tf;
              int tx = ti[qq]; ti[qq] = ti[qq - 1]; ti[qq - 1] = tx;
            }
          }
        }
      }
    }
    __syncthreads();
  }

  if (lane < 16) {
    float mx = tv[0];
    float es[9]; float sum = 0.f;
    #pragma unroll
    for (int t = 0; t < 9; ++t) { es[t] = __expf(tv[t] - mx); sum += es[t]; }
    float inv = 1.0f / sum;
    size_t base = ((size_t)(b * N_) + n0 + lane) * K_;
    #pragma unroll
    for (int t = 0; t < 9; ++t) { topw[base + t] = es[t] * inv; topi[base + t] = ti[t]; }
  }
}

// ---------------- final GEMM: out[b,o,n] = cw2[o,:] . g[b,n,:] + cb[o] -----
// All 8 waves of a block share one 16-row conv-weight slice per tap t; wave 0
// DMAs it into LDS with the Tensor Data Mover (tensor_load_to_lds, 2D D#),
// waits on TENSORcnt, then all waves read A-fragments via ds_load_b128.
// B-fragments are gather+softmax-weighted v rows built on the fly.
__global__ void __launch_bounds__(256) out_gemm(const unsigned short* __restrict__ cw2,
                                                const float* __restrict__ vt,
                                                const float* __restrict__ topw,
                                                const int* __restrict__ topi,
                                                const float* __restrict__ cbias,
                                                float* __restrict__ out) {
  __shared__ unsigned short tileA[16 * C_];   // 8 KB: 16 o-rows x 256 taps (one t)
  int lane = threadIdx.x & 31;
  int wv   = threadIdx.x >> 5;
  int wid = blockIdx.x * 8 + wv;             // B*16*128; ntile fastest => all
  int ntile = wid % (N_ / 16); wid /= (N_ / 16);   // 8 waves share (b,otile)
  int otile = wid % (C_ / 16);
  int b = wid / (C_ / 16);
  int o0 = otile * 16, n0 = ntile * 16;
  int off = (lane < 16) ? 0 : 8;
  int n = n0 + (lane & 15);

  F32Acc acc;
  #pragma unroll
  for (int r = 0; r < 8; ++r) acc.f[r] = cbias[o0 + off + r];

  size_t tkbase = ((size_t)(b * N_) + n) * K_;
  unsigned int ldsaddr = (unsigned int)(size_t)&tileA[0];  // LDS byte offset
  unsigned long long gbase =
      (unsigned long long)(size_t)(cw2 + (size_t)o0 * KK_); // tile row 0, t=0

  for (int t = 0; t < K_; ++t) {
    __syncthreads();                         // previous tile fully consumed
    if (wv == 0) {
      unsigned long long ga = gbase + (unsigned long long)t * C_ * 2;
      // D# group 0: count=1 | lds_addr | global_addr[56:0] | type=2
      u32x4 g0 = {1u, ldsaddr,
                  (unsigned int)ga,
                  (unsigned int)(((ga >> 32) & 0x01FFFFFFu) | 0x80000000u)};
      // D# group 1: data_size=2B; tensor_dim0=2304, dim0_stride=2304,
      //             tensor_dim1=256; tile = 256 x 16
      u32x8 g1 = {0x00010000u,              // data_size=1 (2 bytes)
                  0x09000000u,              // tensor_dim0[15:0]=2304 @ [63:48]
                  0x01000000u,              // tensor_dim1[15:0]=256 @ [95:80]
                  0x01000000u,              // tile_dim0=256        @ [127:112]
                  0x00000010u,              // tile_dim1=16         @ [143:128]
                  0x00000900u,              // dim0_stride=2304     @ [191:160]
                  0u, 0u};
      asm volatile("tensor_load_to_lds %0, %1" :: "s"(g0), "s"(g1) : "memory");
      __builtin_amdgcn_s_wait_tensorcnt(0);
    }
    __syncthreads();                         // tileA valid for all 8 waves

    int   wi = topi[tkbase + t];
    float ww = topw[tkbase + t];
    const float* vrowp = vt + ((size_t)(b * N_) + wi) * C_;
    int cb = (lane < 16) ? 0 : 16;           // this lane's K half of the chunk
    #pragma unroll
    for (int cs = 0; cs < 8; ++cs) {
      int c0 = cs * 32;
      const float* p = vrowp + c0 + cb;
      f32x4 f0 = *(const f32x4*)(p);
      f32x4 f1 = *(const f32x4*)(p + 4);
      f32x4 f2 = *(const f32x4*)(p + 8);
      f32x4 f3 = *(const f32x4*)(p + 12);
      BF16Frag bfv;
      bfv.h[0]  = f2bf(f0.x * ww); bfv.h[1]  = f2bf(f0.y * ww);
      bfv.h[2]  = f2bf(f0.z * ww); bfv.h[3]  = f2bf(f0.w * ww);
      bfv.h[4]  = f2bf(f1.x * ww); bfv.h[5]  = f2bf(f1.y * ww);
      bfv.h[6]  = f2bf(f1.z * ww); bfv.h[7]  = f2bf(f1.w * ww);
      bfv.h[8]  = f2bf(f2.x * ww); bfv.h[9]  = f2bf(f2.y * ww);
      bfv.h[10] = f2bf(f2.z * ww); bfv.h[11] = f2bf(f2.w * ww);
      bfv.h[12] = f2bf(f3.x * ww); bfv.h[13] = f2bf(f3.y * ww);
      bfv.h[14] = f2bf(f3.z * ww); bfv.h[15] = f2bf(f3.w * ww);
      v16bf a = lds_a_frag(tileA, c0, lane);
      acc.v = __builtin_amdgcn_wmma_f32_16x16x32_bf16(false, a, false, bfv.v,
                                                      (short)0, acc.v, false, false);
    }
  }
  float* op = out + ((size_t)(b * C_ + o0 + off)) * N_ + n;
  #pragma unroll
  for (int r = 0; r < 8; ++r) op[(size_t)r * N_] = acc.f[r];
}

// ---------------------------------------------------------------------------
extern "C" void kernel_launch(void* const* d_in, const int* in_sizes, int n_in,
                              void* d_out, int out_size, void* d_ws, size_t ws_size,
                              hipStream_t stream) {
  const float* x    = (const float*)d_in[0];
  const float* wk_w = (const float*)d_in[1];
  const float* wk_b = (const float*)d_in[2];
  const float* wq_w = (const float*)d_in[3];
  const float* wq_b = (const float*)d_in[4];
  const float* wv_w = (const float*)d_in[5];
  const float* wv_b = (const float*)d_in[6];
  const float* cw   = (const float*)d_in[7];
  const float* cb   = (const float*)d_in[8];
  float* outp = (float*)d_out;

  char* ws = (char*)d_ws;
  size_t off = 0;
  auto alloc = [&](size_t bytes) -> void* {
    void* p = ws + off;
    off = (off + bytes + 255) & ~(size_t)255;
    return p;
  };
  unsigned short* x_t  = (unsigned short*)alloc((size_t)B_ * N_ * C_ * 2);  // 16 MiB
  unsigned short* w_bf = (unsigned short*)alloc((size_t)3 * C_ * C_ * 2);
  unsigned short* cw2  = (unsigned short*)alloc((size_t)C_ * KK_ * 2);
  float* vt = (float*)alloc((size_t)B_ * N_ * C_ * 4);                      // 32 MiB
  unsigned short* kn = (unsigned short*)alloc((size_t)B_ * N_ * C_ * 2);    // 16 MiB
  unsigned short* qn = (unsigned short*)alloc((size_t)B_ * N_ * C_ * 2);
  float* topw = (float*)alloc((size_t)B_ * N_ * K_ * 4);
  int*   topi = (int*)alloc((size_t)B_ * N_ * K_ * 4);

  prep_x <<<(B_ * C_ * N_) / 256, 256, 0, stream>>>(x, x_t);
  prep_w <<<(3 * C_ * C_) / 256, 256, 0, stream>>>(wk_w, wq_w, wv_w, w_bf);
  prep_cw<<<(C_ * C_ * K_) / 256, 256, 0, stream>>>(cw, cw2);
  // fused QKV + normalize: one wave per (b,which,ntile)
  qkv_norm<<<B_ * 3 * (N_ / 16), 32, 0, stream>>>(x_t, w_bf, wk_b, wq_b, wv_b,
                                                  kn, qn, vt);
  s_topk<<<B_ * (N_ / 16), 32, 0, stream>>>(kn, qn, topw, topi);
  // 8 waves/block sharing one conv-weight slice staged by TDM
  out_gemm<<<(B_ * (C_ / 16) * (N_ / 16)) / 8, 256, 0, stream>>>(
      cw2, vt, topw, topi, cb, outp);
}
